// GCNLayer_11184094839115
// MI455X (gfx1250) — matrix operand order repaired and checked by reading, not compiled, not run on hardware
//
#include <hip/hip_runtime.h>

// ---------------- problem constants (from reference) ----------------
#define N_NODES 50000
#define N_EDGES 800000
#define D_IN    1024
#define D_OUT   256
#define LEAKY   0.01f

// ---------------- WMMA vector types (gfx1250, wave32) ----------------
typedef __bf16          v16bf  __attribute__((ext_vector_type(16)));
typedef __bf16          bf16x2 __attribute__((ext_vector_type(2)));
typedef float           v8f    __attribute__((ext_vector_type(8)));
typedef unsigned short  u16x8  __attribute__((ext_vector_type(8)));
typedef unsigned short  u16x4  __attribute__((ext_vector_type(4)));

union Frag { v16bf bf; u16x8 h[2]; };

__device__ __forceinline__ unsigned short f2bf_rne(float f) {
    union { float f; unsigned u; } v; v.f = f;
    unsigned r = v.u + 0x7FFFu + ((v.u >> 16) & 1u);
    return (unsigned short)(r >> 16);
}

// Convert 4 floats -> 4 bf16 (packed hw cvt if available on gfx1250)
__device__ __forceinline__ u16x4 cvt4_bf16(float a, float b, float c, float d) {
#if __has_builtin(__builtin_amdgcn_cvt_pk_bf16_f32)
    union { u16x4 u; bf16x2 v[2]; } r;
    r.v[0] = __builtin_amdgcn_cvt_pk_bf16_f32(a, b);
    r.v[1] = __builtin_amdgcn_cvt_pk_bf16_f32(c, d);
    return r.u;
#else
    u16x4 p;
    p.x = f2bf_rne(a); p.y = f2bf_rne(b);
    p.z = f2bf_rne(c); p.w = f2bf_rne(d);
    return p;
#endif
}

// ---------------- one-time: W [1024][256] f32 -> W^T [256][1024] bf16 --
__global__ __launch_bounds__(256) void convert_wT(
    const float* __restrict__ w,          // [D_IN][D_OUT]
    unsigned short* __restrict__ wT)      // [D_OUT][D_IN] bf16 bits
{
    int idx = blockIdx.x * 256 + threadIdx.x;   // 0 .. D_IN*D_OUT-1
    if (idx >= D_IN * D_OUT) return;
    int k = idx >> 8;          // D_OUT == 256
    int n = idx & 255;
    wT[n * D_IN + k] = f2bf_rne(w[idx]);   // coalesced read, strided write (one-time)
}

// ---------------- GEMM: support = x @ W  (bf16 WMMA, f32 accum) -------
// Block tile: 128 (M) x 256 (N), K-chunk 32. 256 threads = 8 waves
// arranged 4 (M) x 2 (N); each wave owns 2 M-tiles x 8 N-tiles of 16x16.
// A staged through double-buffered LDS (f32->bf16 on the fly);
// B fragments loaded from W^T with next-k-step global_prefetch_b8 so the
// unavoidable per-group loadcnt waits hit WGP$ instead of L2.
#define BM   128
#define KC   32
#define SAW  40   // LDS row stride (halfs): 32 + 8 pad, keeps 16B alignment

__global__ __launch_bounds__(256)
void gemm_bf16_wmma(
    const float* __restrict__ x,             // [N_NODES][D_IN] f32
    const unsigned short* __restrict__ wT,   // [D_OUT][D_IN]  bf16 bits
    float* __restrict__ support)             // [N_NODES][D_OUT] f32
{
    __shared__ __align__(16) unsigned short sA[2][BM * SAW];  // 2 x 128 x 32 bf16

    const int tid    = threadIdx.x;
    const int lane   = tid & 31;
    const int wid    = tid >> 5;
    const int waveM  = wid >> 1;      // 0..3
    const int waveN  = wid & 1;       // 0..1
    const int blockM = blockIdx.x * BM;

    const int mRow  = lane & 15;      // fragment row/col within 16
    const int half  = lane >> 4;      // lane-group select

    v8f acc[2][8];
    v8f zero = {};
    #pragma unroll
    for (int i = 0; i < 2; ++i)
        #pragma unroll
        for (int j = 0; j < 8; ++j)
            acc[i][j] = zero;

    // ---- A tile staging: 128 rows x 32 k (4 float4 per thread).
    // OOB rows are clamped (not zeroed): padding A rows only feed padding
    // output rows, which are guarded at the final store. No exec-mask branch.
    auto stageA = [&](int buf, int k0) {
        #pragma unroll
        for (int t = 0; t < 4; ++t) {
            int idx = tid + t * 256;          // 0..1023
            int row = idx >> 3;               // 0..127
            int kv  = idx & 7;                // float4 slot within 32-k chunk
            int gr  = blockM + row;
            gr = (gr < N_NODES) ? gr : (N_NODES - 1);
            float4 v = *(const float4*)(x + (size_t)gr * D_IN + k0 + kv * 4);
            *(u16x4*)(&sA[buf][row * SAW + kv * 4]) = cvt4_bf16(v.x, v.y, v.z, v.w);
        }
    };

    stageA(0, 0);
    __syncthreads();

    int cur = 0;
    for (int k0 = 0; k0 < D_IN; k0 += KC, cur ^= 1) {
        // overlap: stage next A chunk into the other buffer
        if (k0 + KC < D_IN) stageA(cur ^ 1, k0 + KC);

        // ---- A fragments (bf16 16x32 layout): lane(m,half) holds
        //      K = {16*half..16*half+15} (contiguous, matches B)
        Frag a[2];
        #pragma unroll
        for (int i = 0; i < 2; ++i) {
            int aRow = waveM * 32 + i * 16 + mRow;
            a[i].h[0] = *(const u16x8*)(&sA[cur][aRow * SAW + 16 * half]);
            a[i].h[1] = *(const u16x8*)(&sA[cur][aRow * SAW + 16 * half + 8]);
        }

        // ---- B fragments from W^T (L2/WGP$-resident). Layout: lanes 0-15
        //      cover K=k0..k0+15, lanes 16-31 cover k0+16..k0+31, column =
        //      lane&15 -> two contiguous global_load_b128 per fragment.
        //      Prefetch the next k-step's fragment so the following
        //      iteration's loads hit WGP$ instead of L2.
        const bool pf = (k0 + KC) < D_IN;
        #pragma unroll
        for (int j = 0; j < 8; ++j) {
            int col = waveN * 128 + j * 16 + mRow;
            const unsigned short* bp = wT + (size_t)col * D_IN + k0 + half * 16;
            if (pf) __builtin_prefetch(bp + KC, 0, 3);   // global_prefetch_b8
            Frag b;
            b.h[0] = *(const u16x8*)(bp);
            b.h[1] = *(const u16x8*)(bp + 8);
            #pragma unroll
            for (int i = 0; i < 2; ++i) {
                acc[i][j] = __builtin_amdgcn_wmma_f32_16x16x32_bf16(
                    false, a[i].bf, false, b.bf,
                    (short)0, acc[i][j], false, false);
            }
        }

        __syncthreads();
    }

    // ---- store C (f32 16x16 layout: VGPR r -> M = r + 8*(lane>>4))
    #pragma unroll
    for (int i = 0; i < 2; ++i) {
        #pragma unroll
        for (int j = 0; j < 8; ++j) {
            int col = waveN * 128 + j * 16 + mRow;
            #pragma unroll
            for (int r = 0; r < 8; ++r) {
                int row = blockM + waveM * 32 + i * 16 + r + 8 * half;
                if (row < N_NODES)
                    support[(size_t)row * D_OUT + col] = acc[i][j][r];
            }
        }
    }
}

// ---------------- zero init of the accumulation target ----------------
__global__ __launch_bounds__(256) void zero_out(float* __restrict__ out, int n) {
    int i = blockIdx.x * 256 + threadIdx.x;
    if (i < n) out[i] = 0.0f;
}

// ---------------- edge scatter: wave per edge, 8 f32 per lane ---------
__global__ __launch_bounds__(256) void scatter_edges(
    const float* __restrict__ support,
    const long long* __restrict__ edge_index,   // [2][E] int64
    const float* __restrict__ edge_weight,
    float* __restrict__ out, int E)
{
    int gid  = blockIdx.x * 256 + threadIdx.x;
    int e    = gid >> 5;
    int lane = gid & 31;
    if (e >= E) return;

    long long s = edge_index[e];
    long long d = edge_index[(long long)E + e];
    float wgt   = edge_weight[e];

    const float* sp = support + s * (long long)D_OUT + lane * 8;
    float*       op = out     + d * (long long)D_OUT + lane * 8;

    float4 v0 = *(const float4*)(sp);
    float4 v1 = *(const float4*)(sp + 4);

    atomicAdd(op + 0, v0.x * wgt);
    atomicAdd(op + 1, v0.y * wgt);
    atomicAdd(op + 2, v0.z * wgt);
    atomicAdd(op + 3, v0.w * wgt);
    atomicAdd(op + 4, v1.x * wgt);
    atomicAdd(op + 5, v1.y * wgt);
    atomicAdd(op + 6, v1.z * wgt);
    atomicAdd(op + 7, v1.w * wgt);
}

// ---------------- finalize in place: /norm + bias, leaky-ReLU ---------
__global__ __launch_bounds__(256) void finalize(
    float* __restrict__ out,
    const float* __restrict__ norm,   // [N_NODES][1]
    const float* __restrict__ bias,   // [D_OUT]
    int total)
{
    int i = blockIdx.x * 256 + threadIdx.x;
    if (i >= total) return;
    int row = i >> 8;        // D_OUT == 256
    int col = i & 255;
    float v = out[i] / norm[row] + bias[col];
    out[i] = (v > 0.0f) ? v : LEAKY * v;
}

// ---------------- entry point ----------------------------------------
extern "C" void kernel_launch(void* const* d_in, const int* in_sizes, int n_in,
                              void* d_out, int out_size, void* d_ws, size_t ws_size,
                              hipStream_t stream) {
    const float*     x    = (const float*)d_in[0];      // [N_NODES*D_IN]
    const float*     w    = (const float*)d_in[1];      // [D_IN*D_OUT]
    const float*     bias = (const float*)d_in[2];      // [D_OUT]
    const long long* ei   = (const long long*)d_in[3];  // [2*E] int64
    const float*     ew   = (const float*)d_in[4];      // [E]
    const float*     norm = (const float*)d_in[5];      // [N_NODES]
    float*           out  = (float*)d_out;              // [N_NODES*D_OUT]

    float*          sup = (float*)d_ws;                         // [N_NODES*D_OUT] f32
    unsigned short* wT  = (unsigned short*)(sup + (size_t)N_NODES * D_OUT); // [D_OUT*D_IN] bf16

    const int total = N_NODES * D_OUT;

    // 0) one-time W -> W^T bf16 (512 KB, L2-resident for the GEMM)
    convert_wT<<<(D_IN * D_OUT + 255) / 256, 256, 0, stream>>>(w, wT);

    // 1) support = x @ W  (bf16 WMMA, f32 accumulate)
    gemm_bf16_wmma<<<(N_NODES + BM - 1) / BM, 256, 0, stream>>>(x, wT, sup);

    // 2) zero d_out (poisoned by harness; must be re-zeroed deterministically)
    zero_out<<<(total + 255) / 256, 256, 0, stream>>>(out, total);

    // 3) scatter: one wave (32 lanes) per edge
    scatter_edges<<<(N_EDGES * 32 + 255) / 256, 256, 0, stream>>>(sup, ei, ew, out, N_EDGES);

    // 4) out = leaky_relu(out / norm + bias), in place
    finalize<<<(total + 255) / 256, 256, 0, stream>>>(out, norm, bias, total);
}
